// ModuleCorrelation_44547400794827
// MI455X (gfx1250) — matrix operand order repaired and checked by reading, not compiled, not run on hardware
//
#include <hip/hip_runtime.h>

typedef _Float16 v16h __attribute__((ext_vector_type(16)));
typedef float    v8f  __attribute__((ext_vector_type(8)));

#define BB   4
#define CC   64
#define HH   192
#define WW   320
#define MDD  4
#define ND   9              // 2*MD+1
#define HWs  (HH * WW)      // 61440 elements; 245760 B channel stride
#define CHW  (CC * HH * WW)

// Per-wave LDS visibility fence: DS ops from one wave complete in order on
// CDNA5, so cross-lane LDS communication inside a wave only needs a compiler
// code-motion + memory barrier (no s_barrier, no block-wide sync).
static __forceinline__ __device__ void wave_lds_fence() {
  __builtin_amdgcn_wave_barrier();
  asm volatile("" ::: "memory");
}

// One block = 128 threads = 4 fully independent waves; each wave owns 16
// consecutive pixels of one image row. For each dy we compute the banded Gram
// block  G[w2, w1] = sum_c second[c, h+dy-4, w2] * first[c, h, w1]
// with w1 in [w0, w0+16), w2 in [w0-4, w0+28) via 2 M-tiles of
// v_wmma_f32_16x16x32_f16 (K split into 2 chunks of 32 channels), then pull
// the diagonal band out through LDS: out[p, dx] = G[p+dx-4, p] / 64.
__global__ __launch_bounds__(128)
void corr_wmma_f16_kernel(const float* __restrict__ first,
                          const float* __restrict__ second,
                          float* __restrict__ out) {
  __shared__ float Gt[4][32 * 16];   // per-wave 32x16 Gram block (8 KB total)

  const int lane = threadIdx.x & 31;
  const int wave = threadIdx.x >> 5;
  const int lm   = lane & 15;        // M / N index inside the 16x16 tile
  const int h0   = lane >> 4;        // lane half (0/1)
  const int w0   = blockIdx.x * 64 + wave * 16;
  const int h    = blockIdx.y;
  const int b    = blockIdx.z;

  // ---- Preload B fragments: first[b, :, h, w0+lm] as f16 ----
  // 16-bit B 32x16 layout: element e -> K = (e>=8 ? 16 : 0) + h0*8 + (e&7), N = lane%16.
  // Per-chunk base pointer keeps channel offsets within the imm24 range.
  v16h bf[2];
  {
    const float* fp0 = first + (size_t)b * CHW + (size_t)h * WW + (size_t)(w0 + lm);
#pragma unroll
    for (int chunk = 0; chunk < 2; ++chunk) {
      const float* fp = fp0 + (size_t)(chunk * 32) * HWs;
#pragma unroll
      for (int e = 0; e < 16; ++e) {
        const int cl = ((e >> 3) << 4) + (h0 << 3) + (e & 7);   // 0..31 within chunk
        bf[chunk][e] = (_Float16)fp[(size_t)cl * HWs];
      }
    }
  }

  const float scale = 1.0f / (float)CC;
  const size_t outpix = (size_t)h * WW + (size_t)(w0 + lm);

  for (int dy = 0; dy < ND; ++dy) {
    const int h2 = h + dy - MDD;

    if (h2 < 0 || h2 >= HH) {
      // Zero-padded rows: correlation is exactly zero.
#pragma unroll
      for (int i = 0; i < 5; ++i) {
        const int dx = 2 * i + h0;               // lanes 0-15: even, 16-31: odd
        if (dx < ND) {
          const int d = dy * ND + dx;
          __builtin_nontemporal_store(
              0.0f, &out[((size_t)b * (ND * ND) + d) * HWs + outpix]);
        }
      }
      continue;   // uniform across the block
    }

    // Prefetch the next second-row (L2-resident after first touch anyway).
    if (dy + 1 < ND && (h2 + 1) < HH) {
      __builtin_prefetch(&second[(size_t)b * CHW + (size_t)(h2 + 1) * WW + w0], 0, 0);
    }

    // ---- Load ALL four A fragments first (64 unconditional loads in
    // flight), then run the 4 WMMAs. OOB columns are zeroed by a value
    // select (v_cndmask), never by branching around the load. ----
    bool okm[2];
    const float* sp[2];
#pragma unroll
    for (int m = 0; m < 2; ++m) {
      const int  w2  = w0 - MDD + m * 16 + lm;    // this lane's M row
      const bool ok  = (w2 >= 0) && (w2 < WW);
      okm[m] = ok;
      const int  w2c = ok ? w2 : 0;               // clamped, always in-range
      sp[m] = second + (size_t)b * CHW + (size_t)h2 * WW + (size_t)w2c;
    }

    v16h a[2][2];
#pragma unroll
    for (int m = 0; m < 2; ++m) {
#pragma unroll
      for (int chunk = 0; chunk < 2; ++chunk) {
        const float* p = sp[m] + (size_t)(chunk * 32) * HWs;
#pragma unroll
        for (int e = 0; e < 16; ++e) {
          // 16-bit A 16x32 layout: element e -> M = lane%16,
          //   K(within chunk) = (e>=8 ? 16 : 0) + (lane/16)*8 + (e&7)  (0..31)
          const int cl = ((e >> 3) << 4) + (h0 << 3) + (e & 7);
          const float raw = p[(size_t)cl * HWs];  // unconditional load
          a[m][chunk][e] = (_Float16)(okm[m] ? raw : 0.0f);
        }
      }
    }

#pragma unroll
    for (int m = 0; m < 2; ++m) {
      v8f acc = {0.f, 0.f, 0.f, 0.f, 0.f, 0.f, 0.f, 0.f};
      // (neg_a, A, neg_b, B, c_mod, C, reuse_a, reuse_b)
      acc = __builtin_amdgcn_wmma_f32_16x16x32_f16(
          false, a[m][0], false, bf[0], (short)0, acc, false, false);
      acc = __builtin_amdgcn_wmma_f32_16x16x32_f16(
          false, a[m][1], false, bf[1], (short)0, acc, false, false);

      // C/D layout: lane, VGPR r -> row M = r + 8*(lane/16), col N = lane%16.
#pragma unroll
      for (int r = 0; r < 8; ++r) {
        Gt[wave][(m * 16 + h0 * 8 + r) * 16 + lm] = acc[r];
      }
    }

    wave_lds_fence();   // same-wave DS ops are in-order; just stop reordering

    // Band extraction with all 32 lanes: pixel p = lane%16,
    // lanes 0-15 take even dx, lanes 16-31 odd dx.
    // out(p, dx) = G[p + dx, p] / C.   (bank-conflict-free: 17*lm mod 64)
#pragma unroll
    for (int i = 0; i < 5; ++i) {
      const int dx = 2 * i + h0;
      if (dx < ND) {                   // diverges only on the last round
        const float v = Gt[wave][(lm + dx) * 16 + lm] * scale;
        const int d = dy * ND + dx;
        __builtin_nontemporal_store(
            v, &out[((size_t)b * (ND * ND) + d) * HWs + outpix]);
      }
    }

    wave_lds_fence();   // keep next dy's LDS writes behind this dy's reads
  }
}

extern "C" void kernel_launch(void* const* d_in, const int* in_sizes, int n_in,
                              void* d_out, int out_size, void* d_ws, size_t ws_size,
                              hipStream_t stream) {
  (void)in_sizes; (void)n_in; (void)d_ws; (void)ws_size; (void)out_size;
  const float* first  = (const float*)d_in[0];
  const float* second = (const float*)d_in[1];
  float* out = (float*)d_out;

  dim3 block(128, 1, 1);
  dim3 grid(WW / 64, HH, BB);   // 5 x 192 x 4
  hipLaunchKernelGGL(corr_wmma_f16_kernel, grid, block, 0, stream,
                     first, second, out);
}
